// OOModel_15410342658342
// MI455X (gfx1250) — matrix operand order repaired and checked by reading, not compiled, not run on hardware
//
#include <hip/hip_runtime.h>

typedef __bf16 bf16;
typedef __attribute__((ext_vector_type(16))) __bf16 v16bf;
typedef __attribute__((ext_vector_type(8)))  float  v8f;

#define SEQ    1024
#define BATCH  4
#define TOK    4096     // BATCH*SEQ
#define HIDDEN 1024
#define HEADS  16
#define HD     64

__device__ __forceinline__ bf16 f2bf(float f) {
  union { float f; unsigned u; } x; x.f = f;
  unsigned r = (x.u + 0x7FFFu + ((x.u >> 16) & 1u)) >> 16;
  unsigned short h = (unsigned short)r;
  bf16 b; __builtin_memcpy(&b, &h, 2); return b;
}
__device__ __forceinline__ float bf2f(bf16 b) {
  unsigned short h; __builtin_memcpy(&h, &b, 2);
  unsigned u = ((unsigned)h) << 16;
  float f; __builtin_memcpy(&f, &u, 4); return f;
}

// K index inside a 16-bit A-fragment (16x32): element j, lane-half hi.
// ISA 7.12.2: VGPR0..3 hold K={hi*8+0..7}, VGPR4..7 hold K={16+hi*8+0..7}.
__device__ __forceinline__ int a_kidx(int j, int hi) {
  int vg = j >> 1, p = j & 1;
  return ((vg & 4) ? 16 : 0) + hi * 8 + ((vg & 3) << 1) + p;
}

__device__ __forceinline__ v8f wmma_bf16(v16bf a, v16bf b, v8f c) {
  return __builtin_amdgcn_wmma_f32_16x16x32_bf16(false, a, false, b, (short)0, c,
                                                 false, false);
}

// Async global -> LDS copy (16B per lane), tracked with ASYNCcnt.
__device__ __forceinline__ void async_b128(void* lds_dst, const void* gsrc) {
  unsigned off = (unsigned)(uintptr_t)lds_dst;   // low 32 bits = LDS offset
  asm volatile("global_load_async_to_lds_b128 %0, %1, off"
               :: "v"(off), "v"(gsrc) : "memory");
}
__device__ __forceinline__ void wait_async3() {
  asm volatile("s_wait_asynccnt 0x3" ::: "memory");
}
__device__ __forceinline__ void wait_async2() {
  asm volatile("s_wait_asynccnt 0x2" ::: "memory");
}
__device__ __forceinline__ void wait_async0() {
  asm volatile("s_wait_asynccnt 0x0" ::: "memory");
}

// ---------------------------------------------------------------------------
// Pack fp32 row-major [M,K] -> A-fragment-linear bf16 tiles (16x32 per tile).
__global__ void k_pack_a(const float* __restrict__ src, v16bf* __restrict__ dst,
                         int KT, int ld) {
  int tid  = blockIdx.x * blockDim.x + threadIdx.x;
  int lane = tid & 31;
  int tile = tid >> 5;
  int kt = tile % KT, mt = tile / KT;
  int hi = lane >> 4;
  int row = mt * 16 + (lane & 15);
  v16bf v;
#pragma unroll
  for (int j = 0; j < 16; ++j) {
    int k = kt * 32 + a_kidx(j, hi);
    v[j] = f2bf(src[(long)row * ld + k]);
  }
  dst[(long)tile * 32 + lane] = v;
}

// Pack fp32 row-major [K,N] -> B-fragment-linear bf16 tiles (32x16 per tile).
__global__ void k_pack_b(const float* __restrict__ src, v16bf* __restrict__ dst,
                         int NT, int ld) {
  int tid  = blockIdx.x * blockDim.x + threadIdx.x;
  int lane = tid & 31;
  int tile = tid >> 5;
  int nt = tile % NT, kt = tile / NT;
  int hi = lane >> 4, n = nt * 16 + (lane & 15);
  v16bf v;
#pragma unroll
  for (int j = 0; j < 16; ++j)
    v[j] = f2bf(src[(long)(kt * 32 + a_kidx(j, hi)) * ld + n]);
  dst[(long)tile * 32 + lane] = v;
}

// bias (masked) -> A-fragment tiles per batch: rows n (64 tiles), K = m (32 tiles)
__global__ void k_pack_bias(const float* __restrict__ bias,
                            const unsigned char* __restrict__ mask,
                            v16bf* __restrict__ dst) {
  int tid  = blockIdx.x * blockDim.x + threadIdx.x;
  int lane = tid & 31;
  int tile = tid >> 5;                 // ((b*64 + nt)*32 + kt)
  int kt = tile & 31, nt = (tile >> 5) & 63, b = tile >> 11;
  int hi = lane >> 4;
  int n = nt * 16 + (lane & 15);
  v16bf v;
#pragma unroll
  for (int j = 0; j < 16; ++j) {
    int m = kt * 32 + a_kidx(j, hi);
    long idx = ((long)b * SEQ + n) * SEQ + m;
    float x = mask[idx] ? bias[idx] : 0.f;
    v[j] = f2bf(x);
  }
  dst[(long)tile * 32 + lane] = v;
}

// ---------------------------------------------------------------------------
// GEMM1: fused = silu(X @ W_qkvu) ; split into gated / v / q / k (plain bf16)
__global__ __launch_bounds__(256) void k_gemm1(const v16bf* __restrict__ Xa,
                                               const v16bf* __restrict__ Wb,
                                               bf16* __restrict__ gated,
                                               bf16* __restrict__ vp,
                                               bf16* __restrict__ qp,
                                               bf16* __restrict__ kp) {
  const int KT = 32, NT = 384;
  int lane = threadIdx.x & 31, w = threadIdx.x >> 5;
  int wm = w & 3, wn = w >> 2;
  int mt = blockIdx.x * 4 + wm;        // 0..255
  int nb = blockIdx.y * 8 + wn * 4;    // 0..383 (base of 4 n-tiles)
  v8f z = {0.f,0.f,0.f,0.f,0.f,0.f,0.f,0.f};
  v8f acc[4] = {z, z, z, z};
  for (int kt = 0; kt < KT; ++kt) {
    v16bf a = Xa[((long)mt * KT + kt) * 32 + lane];
    int ktp = kt + 2 < KT ? kt + 2 : KT - 1;     // L2 prefetch of B stream
    __builtin_prefetch(&Wb[((long)ktp * NT + nb) * 32 + lane]);
#pragma unroll
    for (int i = 0; i < 4; ++i) {
      v16bf b = Wb[((long)kt * NT + nb + i) * 32 + lane];
      acc[i] = wmma_bf16(a, b, acc[i]);
    }
  }
  int hi = lane >> 4, nc = lane & 15;
#pragma unroll
  for (int i = 0; i < 4; ++i) {
    int col = (nb + i) * 16 + nc;
#pragma unroll
    for (int r = 0; r < 8; ++r) {
      int t = mt * 16 + r + 8 * hi;
      float x = acc[i][r];
      float s = x * __builtin_amdgcn_rcpf(1.f + __expf(-x));   // fast SiLU
      bf16 bv = f2bf(s);
      if (col < 3072)      gated[(long)t * 3072 + col] = bv;
      else if (col < 4096) vp[(long)t * 1024 + (col - 3072)] = bv;
      else if (col < 5120) qp[(long)t * 1024 + (col - 4096)] = bv;
      else                 kp[(long)t * 1024 + (col - 5120)] = bv;
    }
  }
}

// ---------------------------------------------------------------------------
// q (plain) -> B-fragment tiles per (b,h) [kt in d: 2][nt in n: 64], with RoPE twin
__global__ void k_pack_q(const bf16* __restrict__ qp, v16bf* __restrict__ qB,
                         v16bf* __restrict__ qrB) {
  int tid  = blockIdx.x * blockDim.x + threadIdx.x;
  int lane = tid & 31;
  int tile = tid >> 5;                 // ((bh*2 + kt)*64 + nt)
  int nt = tile & 63, kt = (tile >> 6) & 1, bh = tile >> 7;
  int b = bh >> 4, h = bh & 15;
  int hi = lane >> 4;
  int n = nt * 16 + (lane & 15);
  long base = ((long)(b * SEQ + n)) * HIDDEN + h * HD;
  v16bf vq, vqr;
#pragma unroll
  for (int j = 0; j < 16; ++j) {
    int d = kt * 32 + a_kidx(j, hi);
    int i = d >> 1;
    float inv_freq = __expf(-0.2878231366f * (float)i);  // 10000^(-2i/64)
    float x = bf2f(qp[base + d]);
    float y = bf2f(qp[base + (d ^ 1)]);
    float ang = (float)n * inv_freq;
    float c = __cosf(ang), s = __sinf(ang);
    float r = (d & 1) ? (x * c + y * s) : (x * c - y * s);
    vq[j] = f2bf(x);
    vqr[j] = f2bf(r);
  }
  qB[(long)tile * 32 + lane]  = vq;
  qrB[(long)tile * 32 + lane] = vqr;
}

// k (plain) -> A-fragment tiles per (b,h) [mt in m: 64][kt in d: 2], with RoPE twin
__global__ void k_pack_k(const bf16* __restrict__ kp, v16bf* __restrict__ kA,
                         v16bf* __restrict__ krA) {
  int tid  = blockIdx.x * blockDim.x + threadIdx.x;
  int lane = tid & 31;
  int tile = tid >> 5;                 // ((bh*64 + mt)*2 + kt)
  int kt = tile & 1, mt = (tile >> 1) & 63, bh = tile >> 7;
  int b = bh >> 4, h = bh & 15;
  int hi = lane >> 4;
  int m = mt * 16 + (lane & 15);
  long base = ((long)(b * SEQ + m)) * HIDDEN + h * HD;
  v16bf vk, vkr;
#pragma unroll
  for (int j = 0; j < 16; ++j) {
    int d = kt * 32 + a_kidx(j, hi);
    int i = d >> 1;
    float inv_freq = __expf(-0.2878231366f * (float)i);
    float ang = (float)m * inv_freq;
    float c = __cosf(ang), s = __sinf(ang);
    float x = bf2f(kp[base + d]);
    float y = bf2f(kp[base + (d ^ 1)]);
    float r = (d & 1) ? (x * c + y * s) : (x * c - y * s);
    vk[j]  = f2bf(x);
    vkr[j] = f2bf(r);
  }
  kA[(long)tile * 32 + lane]  = vk;
  krA[(long)tile * 32 + lane] = vkr;
}

// v (plain) -> B-fragment tiles per (b,h) [kt in m: 32][nt in d: 4]
__global__ void k_pack_v(const bf16* __restrict__ vp, v16bf* __restrict__ vB) {
  int tid  = blockIdx.x * blockDim.x + threadIdx.x;
  int lane = tid & 31;
  int tile = tid >> 5;                 // ((bh*32 + kt)*4 + nt)
  int nt = tile & 3, kt = (tile >> 2) & 31, bh = tile >> 7;
  int b = bh >> 4, h = bh & 15;
  int hi = lane >> 4, dcol = nt * 16 + (lane & 15);
  v16bf v;
#pragma unroll
  for (int j = 0; j < 16; ++j) {
    int m = kt * 32 + a_kidx(j, hi);
    v[j] = f2bf(vp[((long)(b * SEQ + m)) * HIDDEN + h * HD + dcol]);
  }
  vB[(long)tile * 32 + lane] = v;
}

// ---------------------------------------------------------------------------
// Fused attention. All 8 waves of a block share the same kA/krA/vB tiles, so
// each 12KB m-chunk (4 kA + 4 krA + 4 vB tiles) is DMA'd once into a
// double-buffered LDS stage via global_load_async_to_lds_b128 (ASYNCcnt),
// overlapped with the WMMA work of the previous chunk.
// Stage 1 computes P^T = relu(k.q^T)/S via WMMA; its C fragments map directly
// onto the A fragments of stage 2 (P@V) -- no cross-lane transpose needed.
__global__ __launch_bounds__(256) void k_attn(const v16bf* __restrict__ qB,
                                              const v16bf* __restrict__ qrB,
                                              const v16bf* __restrict__ kA,
                                              const v16bf* __restrict__ krA,
                                              const v16bf* __restrict__ vB,
                                              const v16bf* __restrict__ biasA,
                                              bf16* __restrict__ combined) {
  __shared__ __align__(32) unsigned char sh[2][12288];
  int tid = threadIdx.x;
  int lane = tid & 31, w = tid >> 5;
  int bh = blockIdx.y;
  int b = bh >> 4, h = bh & 15;
  int nt = blockIdx.x * 8 + w;                    // 0..63 n-tiles
  const float inv_len = 1.f / 1024.f;
  long qbase = (long)bh * 128;                    // tiles per (b,h)
  long bbase = (long)b * 2048;                    // bias tiles per batch
  const char* kAb  = (const char*)kA  + (long)bh * 128 * 1024;
  const char* krAb = (const char*)krA + (long)bh * 128 * 1024;
  const char* vBb  = (const char*)vB  + (long)bh * 128 * 1024;

  auto issue = [&](int mc, int buf) {
    char* d = (char*)sh[buf] + tid * 16;
    const char* s0 = kAb  + (long)mc * 4096 + tid * 16;
    const char* s1 = krAb + (long)mc * 4096 + tid * 16;
    const char* s2 = vBb  + (long)mc * 4096 + tid * 16;
    async_b128(d,        s0);
    async_b128(d + 4096, s1);
    async_b128(d + 8192, s2);
  };

  v16bf qb0  = qB [(qbase + 0 * 64 + nt) * 32 + lane];
  v16bf qb1  = qB [(qbase + 1 * 64 + nt) * 32 + lane];
  v16bf qrb0 = qrB[(qbase + 0 * 64 + nt) * 32 + lane];
  v16bf qrb1 = qrB[(qbase + 1 * 64 + nt) * 32 + lane];

  v8f z = {0.f,0.f,0.f,0.f,0.f,0.f,0.f,0.f};
  v8f ap[4] = {z,z,z,z}, ar[4] = {z,z,z,z}, at[4] = {z,z,z,z};

  issue(0, 0);
  issue(1, 1);
  for (int mc = 0; mc < 32; ++mc) {
    if (mc < 31) wait_async3(); else wait_async0();  // chunk mc landed
    __syncthreads();
    const char* S = (const char*)sh[mc & 1];
    v16bf pa, pra;
#pragma unroll
    for (int hf = 0; hf < 2; ++hf) {
      v16bf ka0 = *(const v16bf*)(S +        (hf * 2 + 0) * 1024 + lane * 32);
      v16bf ka1 = *(const v16bf*)(S +        (hf * 2 + 1) * 1024 + lane * 32);
      v16bf kr0 = *(const v16bf*)(S + 4096 + (hf * 2 + 0) * 1024 + lane * 32);
      v16bf kr1 = *(const v16bf*)(S + 4096 + (hf * 2 + 1) * 1024 + lane * 32);
      v8f c = z, cr = z;
      c  = wmma_bf16(ka0, qb0, c);
      c  = wmma_bf16(ka1, qb1, c);
      cr = wmma_bf16(kr0, qrb0, cr);
      cr = wmma_bf16(kr1, qrb1, cr);
#pragma unroll
      for (int r = 0; r < 8; ++r) {
        float x = c[r];   x  = x  > 0.f ? x  * inv_len : 0.f;
        float xr = cr[r]; xr = xr > 0.f ? xr * inv_len : 0.f;
        pa [hf * 8 + r] = f2bf(x);
        pra[hf * 8 + r] = f2bf(xr);
      }
    }
    v16bf tb = biasA[(bbase + (long)nt * 32 + mc) * 32 + lane];
#pragma unroll
    for (int dt = 0; dt < 4; ++dt) {
      v16bf vb = *(const v16bf*)(S + 8192 + dt * 1024 + lane * 32);
      ap[dt] = wmma_bf16(pa,  vb, ap[dt]);
      ar[dt] = wmma_bf16(pra, vb, ar[dt]);
      at[dt] = wmma_bf16(tb,  vb, at[dt]);
    }
    __syncthreads();                 // everyone done reading sh[mc&1]
    if (mc + 2 < 32) issue(mc + 2, mc & 1);
  }

  // combined[b,s, h*192 + {rope:0, ts:64, plain:128} + d]
  int hi = lane >> 4, nc = lane & 15;
#pragma unroll
  for (int dt = 0; dt < 4; ++dt) {
#pragma unroll
    for (int r = 0; r < 8; ++r) {
      int s = nt * 16 + r + 8 * hi;
      long t = (long)b * SEQ + s;
      int d = dt * 16 + nc;
      long rowoff = t * 3072 + (long)h * 192;
      combined[rowoff + 0   + d] = f2bf(ar[dt][r]);  // rope_out
      combined[rowoff + 64  + d] = f2bf(at[dt][r]);  // ts_out
      combined[rowoff + 128 + d] = f2bf(ap[dt][r]);  // plain_out
    }
  }
}

// combined * gated -> A-fragment tiles (M=4096, K=3072, KT=96)
__global__ void k_pack_cg(const bf16* __restrict__ combined,
                          const bf16* __restrict__ gated,
                          v16bf* __restrict__ dst) {
  int tid  = blockIdx.x * blockDim.x + threadIdx.x;
  int lane = tid & 31;
  int tile = tid >> 5;
  int kt = tile % 96, mt = tile / 96;
  int hi = lane >> 4;
  int row = mt * 16 + (lane & 15);
  v16bf v;
#pragma unroll
  for (int j = 0; j < 16; ++j) {
    int k = kt * 32 + a_kidx(j, hi);
    long idx = (long)row * 3072 + k;
    v[j] = f2bf(bf2f(combined[idx]) * bf2f(gated[idx]));
  }
  dst[(long)tile * 32 + lane] = v;
}

// ---------------------------------------------------------------------------
// GEMM3: out = rmsnorm((cg @ W_out) + b_out + hidden) * rms_w
// One block = one 16-token row tile x full 1024 cols. The A fragments are
// shared by all 8 waves -> stage them through async-LDS double buffers
// (8 kt per 8KB chunk). Norm fused via an LDS ds_add reduction.
__global__ __launch_bounds__(256) void k_gemm3(const v16bf* __restrict__ cg,
                                               const v16bf* __restrict__ Wo,
                                               const float* __restrict__ hidden,
                                               const float* __restrict__ b_out,
                                               const float* __restrict__ rms_w,
                                               float* __restrict__ out) {
  __shared__ __align__(32) unsigned char shA[2][8192];
  __shared__ float ssq[16];
  __shared__ float scl[16];
  const int NT = 64;
  int tid = threadIdx.x;
  int lane = tid & 31, w = tid >> 5;
  int mt = blockIdx.x;
  const char* cgb = (const char*)cg + (long)mt * 96 * 1024;

  auto issue = [&](int c, int buf) {
    char* d = (char*)shA[buf] + tid * 32;
    const char* s = cgb + (long)c * 8192 + tid * 32;
    async_b128(d,      s);
    async_b128(d + 16, s + 16);
  };

  v8f z = {0.f,0.f,0.f,0.f,0.f,0.f,0.f,0.f};
  v8f acc[8] = {z,z,z,z,z,z,z,z};
  issue(0, 0);
  issue(1, 1);
  for (int c = 0; c < 12; ++c) {       // 12 chunks x 8 kt
    if (c < 11) wait_async2(); else wait_async0();
    __syncthreads();
    const char* S = (const char*)shA[c & 1];
    for (int kk = 0; kk < 8; ++kk) {
      int kt = c * 8 + kk;
      v16bf a = *(const v16bf*)(S + kk * 1024 + lane * 32);
#pragma unroll
      for (int i = 0; i < 8; ++i) {
        int ntl = w * 8 + i;
        v16bf b = Wo[((long)kt * NT + ntl) * 32 + lane];
        acc[i] = wmma_bf16(a, b, acc[i]);
      }
    }
    __syncthreads();
    if (c + 2 < 12) issue(c + 2, c & 1);
  }

  if (tid < 16) ssq[tid] = 0.f;
  __syncthreads();
  int hi = lane >> 4, nc = lane & 15;
  float lsq[8] = {0.f,0.f,0.f,0.f,0.f,0.f,0.f,0.f};
#pragma unroll
  for (int i = 0; i < 8; ++i) {
    int col = (w * 8 + i) * 16 + nc;
#pragma unroll
    for (int r = 0; r < 8; ++r) {
      int t = mt * 16 + r + 8 * hi;
      float res = acc[i][r] + b_out[col] + hidden[(long)t * 1024 + col];
      acc[i][r] = res;
      lsq[r] += res * res;
    }
  }
#pragma unroll
  for (int r = 0; r < 8; ++r) atomicAdd(&ssq[hi * 8 + r], lsq[r]);
  __syncthreads();
  if (tid < 16)
    scl[tid] = rsqrtf(ssq[tid] * (1.f / 1024.f) + 1e-6f);
  __syncthreads();
#pragma unroll
  for (int i = 0; i < 8; ++i) {
    int col = (w * 8 + i) * 16 + nc;
    float rw = rms_w[col];
#pragma unroll
    for (int r = 0; r < 8; ++r) {
      int t = mt * 16 + r + 8 * hi;
      out[(long)t * 1024 + col] = acc[i][r] * scl[r + 8 * hi] * rw;
    }
  }
}

// ---------------------------------------------------------------------------
extern "C" void kernel_launch(void* const* d_in, const int* in_sizes, int n_in,
                              void* d_out, int out_size, void* d_ws, size_t ws_size,
                              hipStream_t stream) {
  const float* hidden        = (const float*)d_in[0];
  const unsigned char* mask  = (const unsigned char*)d_in[1];
  const float* bias          = (const float*)d_in[2];
  const float* Wq            = (const float*)d_in[3];
  const float* Wo            = (const float*)d_in[4];
  const float* b_out         = (const float*)d_in[5];
  const float* rms_w         = (const float*)d_in[6];
  float* out = (float*)d_out;

  char* ws = (char*)d_ws;
  size_t off = 0;
  auto alloc = [&](size_t bytes) -> char* {
    char* p = ws + off;
    off = (off + bytes + 255) & ~(size_t)255;
    return p;
  };
  v16bf* Xa   = (v16bf*)alloc((size_t)TOK * HIDDEN * 2);        // X A-pack
  v16bf* Wqp  = (v16bf*)alloc((size_t)HIDDEN * 6144 * 2);       // W_qkvu B-pack
  v16bf* Wop  = (v16bf*)alloc((size_t)3072 * 1024 * 2);         // W_out B-pack
  v16bf* Bp   = (v16bf*)alloc((size_t)BATCH * SEQ * SEQ * 2);   // bias A-pack
  bf16*  gated= (bf16*) alloc((size_t)TOK * 3072 * 2);
  bf16*  qp   = (bf16*) alloc((size_t)TOK * 1024 * 2);
  bf16*  kp   = (bf16*) alloc((size_t)TOK * 1024 * 2);
  bf16*  vp   = (bf16*) alloc((size_t)TOK * 1024 * 2);
  v16bf* qBp  = (v16bf*)alloc((size_t)TOK * 1024 * 2);
  v16bf* qrBp = (v16bf*)alloc((size_t)TOK * 1024 * 2);
  v16bf* kAp  = (v16bf*)alloc((size_t)TOK * 1024 * 2);
  v16bf* krAp = (v16bf*)alloc((size_t)TOK * 1024 * 2);
  v16bf* vBp  = (v16bf*)alloc((size_t)TOK * 1024 * 2);
  bf16*  comb = (bf16*) alloc((size_t)TOK * 3072 * 2);
  v16bf* cgp  = (v16bf*)alloc((size_t)TOK * 3072 * 2);

  // pack inputs
  k_pack_a   <<<1024, 256, 0, stream>>>(hidden, Xa, 32, 1024);
  k_pack_b   <<<1536, 256, 0, stream>>>(Wq, Wqp, 384, 6144);
  k_pack_b   <<< 768, 256, 0, stream>>>(Wo, Wop, 64, 1024);
  k_pack_bias<<<1024, 256, 0, stream>>>(bias, mask, Bp);
  // fused QKVU GEMM + SiLU + split
  k_gemm1<<<dim3(64, 48), 256, 0, stream>>>(Xa, Wqp, gated, vp, qp, kp);
  // head packing + RoPE
  k_pack_q<<<1024, 256, 0, stream>>>(qp, qBp, qrBp);
  k_pack_k<<<1024, 256, 0, stream>>>(kp, kAp, krAp);
  k_pack_v<<<1024, 256, 0, stream>>>(vp, vBp);
  // fused attention (plain + rope + ts-bias paths), async-LDS staged
  k_attn<<<dim3(8, 64), 256, 0, stream>>>(qBp, qrBp, kAp, krAp, vBp, Bp, comb);
  // gating + output GEMM + residual + RMSNorm
  k_pack_cg<<<3072, 256, 0, stream>>>(comb, gated, cgp);
  k_gemm3<<<256, 256, 0, stream>>>(cgp, Wop, hidden, b_out, rms_w, out);
}
